// ImportOperator_83210696393485
// MI455X (gfx1250) — 8-bit matrix operand order repaired and checked by reading, compiled, not run on hardware
//
#include <hip/hip_runtime.h>
#include <stdint.h>


typedef int   v8i __attribute__((ext_vector_type(8)));
typedef float v8f __attribute__((ext_vector_type(8)));
typedef float v2f __attribute__((ext_vector_type(2)));

#define M_TOT 65536
#define K_DIM 1024
#define N_DIM 1024
#define G_SZ  128
#define NG    8

__global__ __launch_bounds__(256)
void quant_x_kernel(const float* __restrict__ x,
                    const float* __restrict__ in_scale,
                    const int*   __restrict__ in_zp,
                    int8_t* __restrict__ qx,
                    int*    __restrict__ rsum) {
    const int row  = blockIdx.x;
    const int wv   = threadIdx.x >> 5;
    const int lane = threadIdx.x & 31;

    const float s   = in_scale[0];
    const float inv = 1.0f / s;
    const float zpf = (float)in_zp[0];
    const int   zpi = in_zp[0];

    const int k0 = wv * G_SZ + lane * 4;
    const float4 xv = *(const float4*)(x + (size_t)row * K_DIM + k0);

    float f[4] = {xv.x, xv.y, xv.z, xv.w};
    int packed = 0, ssum = 0;
#pragma unroll
    for (int i = 0; i < 4; ++i) {
        float qf = rintf(f[i] * inv) + zpf;
        qf = fminf(fmaxf(qf, -128.0f), 127.0f);
        int qc = (int)qf - zpi;
        qc = max(-128, min(127, qc));
        packed |= (qc & 0xFF) << (8 * i);
        ssum += qc;
    }
    *(int*)(qx + (size_t)row * K_DIM + k0) = packed;

#pragma unroll
    for (int off = 16; off > 0; off >>= 1)
        ssum += __shfl_xor(ssum, off, 32);
    if (lane == 0) rsum[row * NG + wv] = ssum;
}

__global__ __launch_bounds__(256)
void pack_w_kernel(const int* __restrict__ w_q, uint8_t* __restrict__ w8) {
    const int idx = blockIdx.x * 256 + threadIdx.x;
    int4 c = ((const int4*)w_q)[idx];
    ((int*)w8)[idx] = (c.x & 0xFF) | ((c.y & 0xFF) << 8) |
                      ((c.z & 0xFF) << 16) | ((c.w & 0xFF) << 24);
}

__global__ __launch_bounds__(256)
void gemm_iu8_kernel(const int8_t*  __restrict__ qx,
                     const int*     __restrict__ rsum,
                     const uint8_t* __restrict__ w8,
                     const float*   __restrict__ w_scale,
                     const int*     __restrict__ w_zp,
                     const int*     __restrict__ bias_q,
                     const float*   __restrict__ bias_scale,
                     const float*   __restrict__ in_scale,
                     const float*   __restrict__ out_scale,
                     const int*     __restrict__ out_zp,
                     float* __restrict__ out) {
    const int tid    = threadIdx.x;
    const int wv     = tid >> 5;
    const int lane   = tid & 31;
    const int lane15 = lane & 15;
    const int half   = lane >> 4;

    const int mBlock = blockIdx.y * 128;
    const int nBlock = blockIdx.x * 128;
    const int waveM  = (wv >> 2) * 64;
    const int waveN  = (wv & 3) * 32;

    const float sIn = in_scale[0];

    int nc[2];
    nc[0] = nBlock + waveN + lane15;
    nc[1] = nc[0] + 16;

    v2f rfrag[4][2];
#pragma unroll
    for (int mt = 0; mt < 4; ++mt) {
        const int row = mBlock + waveM + mt * 16 + lane15;
#pragma unroll
        for (int ks = 0; ks < 2; ++ks) {
            const int g0 = ks * 4 + half * 2;
            rfrag[mt][ks][0] = (float)rsum[(size_t)row * NG + g0];
            rfrag[mt][ks][1] = (float)rsum[(size_t)row * NG + g0 + 1];
        }
    }
    v2f bsfrag[2][2];
#pragma unroll
    for (int nt = 0; nt < 2; ++nt) {
        const size_t nb = (size_t)nc[nt] * NG;
#pragma unroll
        for (int ks = 0; ks < 2; ++ks) {
            const int g0 = ks * 4 + half * 2;
            bsfrag[nt][ks][0] = -sIn * w_scale[nb + g0]     * (float)w_zp[nb + g0];
            bsfrag[nt][ks][1] = -sIn * w_scale[nb + g0 + 1] * (float)w_zp[nb + g0 + 1];
        }
    }
    v8f accf[8];
#pragma unroll
    for (int mt = 0; mt < 4; ++mt)
#pragma unroll
        for (int nt = 0; nt < 2; ++nt) {
            v8f z = {};
            v8f c0 = __builtin_amdgcn_wmma_f32_16x16x4_f32(
                false, rfrag[mt][0], false, bsfrag[nt][0], (short)0, z, false, false);
            accf[mt * 2 + nt] = __builtin_amdgcn_wmma_f32_16x16x4_f32(
                false, rfrag[mt][1], false, bsfrag[nt][1], (short)0, c0, false, false);
        }

    const int8_t* aRow[4];
#pragma unroll
    for (int mt = 0; mt < 4; ++mt)
        aRow[mt] = qx + (size_t)(mBlock + waveM + mt * 16 + lane15) * K_DIM + half * 16;
    const uint8_t* bRow[2];
#pragma unroll
    for (int nt = 0; nt < 2; ++nt)
        bRow[nt] = w8 + (size_t)nc[nt] * K_DIM + half * 16;

    for (int g = 0; g < NG; ++g) {
        v8i ci[8];
#pragma unroll
        for (int t = 0; t < 8; ++t) { v8i z = {}; ci[t] = z; }

#pragma unroll
        for (int ks = 0; ks < 2; ++ks) {
            const int kk = g * G_SZ + ks * 64;

            v8i a[4];
#pragma unroll
            for (int mt = 0; mt < 4; ++mt) {
                const int8_t* p = aRow[mt] + kk;
                int2 d0 = *(const int2*)(p);
                int2 d1 = *(const int2*)(p + 8);
                int2 d2 = *(const int2*)(p + 32);
                int2 d3 = *(const int2*)(p + 40);
                v8i t_;
                t_[0] = d0.x; t_[1] = d0.y; t_[2] = d1.x; t_[3] = d1.y;
                t_[4] = d2.x; t_[5] = d2.y; t_[6] = d3.x; t_[7] = d3.y;
                a[mt] = t_;
            }
            v8i b[2];
#pragma unroll
            for (int nt = 0; nt < 2; ++nt) {
                const uint8_t* p = bRow[nt] + kk;
                int4 e0 = *(const int4*)(p);
                int4 e1 = *(const int4*)(p + 32);
                v8i t_;
                t_[0] = e0.x; t_[1] = e0.y; t_[2] = e0.z; t_[3] = e0.w;
                t_[4] = e1.x; t_[5] = e1.y; t_[6] = e1.z; t_[7] = e1.w;
                b[nt] = t_;
            }

#pragma unroll
            for (int mt = 0; mt < 4; ++mt)
#pragma unroll
                for (int nt = 0; nt < 2; ++nt)
                    ci[mt * 2 + nt] = __builtin_amdgcn_wmma_i32_16x16x64_iu8(
                         true, a[mt],  false, b[nt],
                        ci[mt * 2 + nt],  false,  false);
        }

#pragma unroll
        for (int nt = 0; nt < 2; ++nt) {
            const float sg = sIn * w_scale[(size_t)nc[nt] * NG + g];
#pragma unroll
            for (int mt = 0; mt < 4; ++mt) {
                const int t = mt * 2 + nt;
#pragma unroll
                for (int v = 0; v < 8; ++v)
                    accf[t][v] += sg * (float)ci[t][v];
            }
        }
    }

    const float sOut   = out_scale[0];
    const float invOut = 1.0f / sOut;
    const float zOut   = (float)out_zp[0];
#pragma unroll
    for (int nt = 0; nt < 2; ++nt) {
        const int n = nc[nt];
        const float bias = ((float)bias_q[n] - 128.0f) * bias_scale[n];
#pragma unroll
        for (int mt = 0; mt < 4; ++mt) {
#pragma unroll
            for (int v = 0; v < 8; ++v) {
                float val = accf[mt * 2 + nt][v] + bias;
                float qf  = rintf(val * invOut) + zOut;
                qf = fminf(fmaxf(qf, -128.0f), 127.0f);
                const float o = (qf - zOut) * sOut;
                const int gm = mBlock + waveM + mt * 16 + v + half * 8;
                out[(size_t)gm * N_DIM + n] = o;
            }
        }
    }
}

extern "C" void kernel_launch(void* const* d_in, const int* in_sizes, int n_in,
                              void* d_out, int out_size, void* d_ws, size_t ws_size,
                              hipStream_t stream) {
    (void)in_sizes; (void)n_in; (void)out_size; (void)ws_size;

    const float* x          = (const float*)d_in[0];
    const int*   w_q        = (const int*)d_in[1];
    const float* w_scale    = (const float*)d_in[2];
    const int*   w_zp       = (const int*)d_in[3];
    const int*   bias_q     = (const int*)d_in[4];
    const float* bias_scale = (const float*)d_in[5];
    const float* in_scale   = (const float*)d_in[6];
    const int*   in_zp      = (const int*)d_in[7];
    const float* out_scale  = (const float*)d_in[8];
    const int*   out_zp     = (const int*)d_in[9];
    float*       out        = (float*)d_out;

    int8_t*  qx   = (int8_t*)d_ws;
    int*     rsum = (int*)((char*)d_ws + (size_t)M_TOT * K_DIM);
    uint8_t* w8   = (uint8_t*)((char*)d_ws + (size_t)M_TOT * K_DIM
                               + (size_t)M_TOT * NG * sizeof(int));

    quant_x_kernel<<<dim3(M_TOT), 256, 0, stream>>>(x, in_scale, in_zp, qx, rsum);
    pack_w_kernel<<<dim3((N_DIM * K_DIM / 4) / 256), 256, 0, stream>>>(w_q, w8);
    gemm_iu8_kernel<<<dim3(N_DIM / 128, M_TOT / 128), 256, 0, stream>>>(
        qx, rsum, w8, w_scale, w_zp, bias_q, bias_scale,
        in_scale, out_scale, out_zp, out);
}
